// Decoder_88605175316972
// MI455X (gfx1250) — compile-verified, matrix operand order checked by reading, not run on hardware
//
#include <hip/hip_runtime.h>
#include <hip/hip_bf16.h>
#include <math.h>

// ---------------------------------------------------------------------------
// Transformer decoder forward for MI455X (gfx1250, wave32).
// All GEMMs run through v_wmma_f32_16x16x32_bf16 (bf16 inputs, f32 accum).
// Tile staging uses async global->LDS copies (ASYNCcnt) when available.
// ---------------------------------------------------------------------------

typedef unsigned short u16;
typedef float  v8f  __attribute__((ext_vector_type(8)));
typedef __bf16 v16bf __attribute__((ext_vector_type(16)));
typedef unsigned int v4u __attribute__((ext_vector_type(4)));
typedef int v4i __attribute__((ext_vector_type(4)));

// Async global->LDS path (gfx1250): gate on builtin availability so the file
// still compiles (with the synchronous fallback) on toolchains without them.
#if defined(__AMDGCN__) && \
    __has_builtin(__builtin_amdgcn_global_load_async_to_lds_b128) && \
    __has_builtin(__builtin_amdgcn_s_wait_asynccnt)
#define USE_ASYNC_LDS 1
#define ASG __attribute__((address_space(1)))
#define ASL __attribute__((address_space(3)))
#else
#define USE_ASYNC_LDS 0
#endif

// Problem constants (match reference)
constexpr int VOCAB = 32000;
constexpr int DM    = 512;
constexpr int NL    = 2;
constexpr int NH    = 8;
constexpr int NB    = 4;
constexpr int TSRC  = 512;
constexpr int TTGT  = 512;
constexpr int HD    = NH * DM;       // 4096
constexpr int BT    = NB * TTGT;     // 2048 target rows
constexpr int BS    = NB * TSRC;     // 2048 source rows
constexpr int FF    = 4 * DM;        // 2048
constexpr int NV    = VOCAB + 1;     // 32001

__device__ __forceinline__ u16 f2bf(float f) {
  unsigned u = __float_as_uint(f);
  unsigned r = u + 0x7FFFu + ((u >> 16) & 1u);   // round-to-nearest-even
  return (u16)(r >> 16);
}

// ---------------------------------------------------------------------------
// Generic batched GEMM:  C[z] = alpha * A[z] x op(B[z]) + bias  (opt. ReLU)
// A: bf16 [M,K] row-major.  B: bf16, NN=[K,N] row-major, NT=[N,K] row-major.
// Batch z decomposes as (bz, hz) with z = bz*batchH + hz; per-dim strides.
// Block tile 128x128, K-step 32. 256 threads = 8 waves; wave grid 2x4,
// each wave computes a 64x32 patch = 4x2 tiles of 16x16 via WMMA.
// ---------------------------------------------------------------------------
#define BMT 128
#define BNT 128
#define BKT 32
#define LDT 40   // LDS stride in halfs (16B aligned rows, conflict padding)

union Frag { v4u u[2]; v16bf v; };

__global__ __launch_bounds__(256)
void gemm_bf16_wmma(const u16* __restrict__ A, long long lda, long long sAb, long long sAh,
                    const u16* __restrict__ Bm, long long ldb, long long sBb, long long sBh,
                    int transB,
                    float* __restrict__ C, long long ldc, long long sCb, long long sCh,
                    int M, int N, int K, int batchH,
                    const float* __restrict__ bias, float alpha, int relu)
{
  __shared__ u16 sA[BMT * LDT];
  __shared__ u16 sB[BNT * LDT];

  const int z  = blockIdx.z;
  const int bz = z / batchH;
  const int hz = z - bz * batchH;
  A  += (size_t)bz * sAb + (size_t)hz * sAh;
  Bm += (size_t)bz * sBb + (size_t)hz * sBh;
  C  += (size_t)bz * sCb + (size_t)hz * sCh;

  const int bm0 = blockIdx.y * BMT;
  const int bn0 = blockIdx.x * BNT;
  const int tid  = threadIdx.x;
  const int lane = tid & 31;
  const int wave = tid >> 5;
  const int mw   = wave >> 2;   // 0..1
  const int nw   = wave & 3;    // 0..3

  v8f acc[4][2];
  const v8f z8 = {0.f,0.f,0.f,0.f,0.f,0.f,0.f,0.f};
#pragma unroll
  for (int mi = 0; mi < 4; ++mi)
#pragma unroll
    for (int ni = 0; ni < 2; ++ni) acc[mi][ni] = z8;

  const v4u zero4 = {0u,0u,0u,0u};

  for (int k0 = 0; k0 < K; k0 += BKT) {
    // ---- stage A tile: 128 rows x 32 halfs; 2 threads/row, 16 halfs each
    {
      const int r    = tid >> 1;
      const int part = (tid & 1) * 16;
      const int grow = bm0 + r;
      const u16* ga = A + (size_t)grow * lda + k0 + part;
      u16* la = &sA[r * LDT + part];
      if (grow < M) {
#if USE_ASYNC_LDS
        // async DMA: LDS[la+off] = GM[ga+off]; VGPRs bypassed, ASYNCcnt tracked
        __builtin_amdgcn_global_load_async_to_lds_b128(
            (ASG v4i*)ga, (ASL v4i*)la, 0, 0);
        __builtin_amdgcn_global_load_async_to_lds_b128(
            (ASG v4i*)ga, (ASL v4i*)la, 16, 0);
#else
        const v4u* gv = (const v4u*)ga;
        *(v4u*)la       = gv[0];
        *(v4u*)(la + 8) = gv[1];
#endif
        if (k0 + BKT < K)  // CDNA5 global_prefetch_b8 for next K tile
          __builtin_prefetch(A + (size_t)grow * lda + k0 + BKT, 0, 1);
      }
#if !USE_ASYNC_LDS
      else {
        *(v4u*)la       = zero4;
        *(v4u*)(la + 8) = zero4;
      }
#endif
    }
    // ---- stage B tile column-major in LDS: sB[col][k], stride LDT
    if (transB == 0) {
      // B is [K,N] row-major; scalar guarded loads (handles N=32001, odd ldb)
      const int kk = tid >> 3;           // 0..31
      const int cb = (tid & 7) * 16;     // 0..112
      const u16* gb = Bm + (size_t)(k0 + kk) * ldb + bn0 + cb;
#pragma unroll
      for (int j = 0; j < 16; ++j) {
        const int col = cb + j;
        u16 v = (bn0 + col < N) ? gb[j] : (u16)0;
        sB[col * LDT + kk] = v;
      }
    } else {
      // B is [N,K] row-major (computes A x B^T); vectorized
      const int cc   = tid >> 1;
      const int part = (tid & 1) * 16;
      const int grow = bn0 + cc;
      const u16* gb = Bm + (size_t)grow * ldb + k0 + part;
      u16* lb = &sB[cc * LDT + part];
      if (grow < N) {
#if USE_ASYNC_LDS
        __builtin_amdgcn_global_load_async_to_lds_b128(
            (ASG v4i*)gb, (ASL v4i*)lb, 0, 0);
        __builtin_amdgcn_global_load_async_to_lds_b128(
            (ASG v4i*)gb, (ASL v4i*)lb, 16, 0);
#else
        const v4u* gv = (const v4u*)gb;
        *(v4u*)lb       = gv[0];
        *(v4u*)(lb + 8) = gv[1];
#endif
      }
#if !USE_ASYNC_LDS
      else {
        *(v4u*)lb       = zero4;
        *(v4u*)(lb + 8) = zero4;
      }
#endif
    }
#if USE_ASYNC_LDS
    __builtin_amdgcn_s_wait_asynccnt(0);   // async LDS writes visible
#endif
    __syncthreads();

    // ---- WMMA fragments (ISA 7.12.2): lane&15 = row(M)/col(N),
    //      lane>>4 selects K-half; halves 0..7 = K kb..kb+7, 8..15 = kb+16..kb+23
    const int r  = lane & 15;
    const int kb = (lane >> 4) * 8;
    Frag fa[4], fb[2];
#pragma unroll
    for (int mi = 0; mi < 4; ++mi) {
      const int row = mw * 64 + mi * 16 + r;
      fa[mi].u[0] = *(const v4u*)&sA[row * LDT + kb];
      fa[mi].u[1] = *(const v4u*)&sA[row * LDT + kb + 16];
    }
#pragma unroll
    for (int ni = 0; ni < 2; ++ni) {
      const int col = nw * 32 + ni * 16 + r;
      fb[ni].u[0] = *(const v4u*)&sB[col * LDT + kb];
      fb[ni].u[1] = *(const v4u*)&sB[col * LDT + kb + 16];
    }
#pragma unroll
    for (int mi = 0; mi < 4; ++mi)
#pragma unroll
      for (int ni = 0; ni < 2; ++ni)
        acc[mi][ni] = __builtin_amdgcn_wmma_f32_16x16x32_bf16(
            false, fa[mi].v, false, fb[ni].v, (short)0, acc[mi][ni], false, false);
    __syncthreads();
  }

  // ---- epilogue: D layout = lane&15 col, (lane>>4)*8 + r rows in 8 VGPRs
#pragma unroll
  for (int mi = 0; mi < 4; ++mi) {
#pragma unroll
    for (int ni = 0; ni < 2; ++ni) {
      const int gcol = bn0 + nw * 32 + ni * 16 + (lane & 15);
      const int grb  = bm0 + mw * 64 + mi * 16 + (lane >> 4) * 8;
      if (gcol < N) {
        const float bv = bias ? bias[gcol] : 0.f;
#pragma unroll
        for (int rr = 0; rr < 8; ++rr) {
          const int grow = grb + rr;
          if (grow < M) {
            float v = acc[mi][ni][rr] * alpha + bv;
            if (relu) v = fmaxf(v, 0.f);
            C[(size_t)grow * ldc + gcol] = v;
          }
        }
      }
    }
  }
  (void)zero4;
}

// ---------------------------------------------------------------------------
// Elementwise / reduction kernels
// ---------------------------------------------------------------------------
__global__ void f32_to_bf16_kernel(const float* __restrict__ s, u16* __restrict__ d,
                                   long long n) {
  long long i = (long long)blockIdx.x * blockDim.x + threadIdx.x;
  if (i < n) d[i] = f2bf(s[i]);
}

// Wq[h][d][e] (f32) -> wbf[d][h*DM+e] (bf16)
__global__ void permute_qkv_w_kernel(const float* __restrict__ src, u16* __restrict__ dst) {
  long long i = (long long)blockIdx.x * blockDim.x + threadIdx.x;
  if (i >= (long long)DM * HD) return;
  int col = (int)(i % HD);
  int d   = (int)(i / HD);
  int h   = col / DM, e = col % DM;
  dst[i] = f2bf(src[((size_t)h * DM + d) * DM + e]);
}

__global__ void pad_from_tokens_kernel(const int* __restrict__ t, float* __restrict__ p,
                                       int n, int padval) {
  int i = blockIdx.x * blockDim.x + threadIdx.x;
  if (i < n) p[i] = (t[i] != padval) ? 1.f : 0.f;
}
__global__ void pad_from_int_kernel(const int* __restrict__ s, float* __restrict__ p, int n) {
  int i = blockIdx.x * blockDim.x + threadIdx.x;
  if (i < n) p[i] = (s[i] != 0) ? 1.f : 0.f;
}

// x = emb[target] + sinusoidal positional encoding
__global__ void embed_kernel(const int* __restrict__ target, const float* __restrict__ emb,
                             float* __restrict__ xf, u16* __restrict__ xbf) {
  long long i = (long long)blockIdx.x * blockDim.x + threadIdx.x;
  if (i >= (long long)BT * DM) return;
  int d   = (int)(i % DM);
  int row = (int)(i / DM);
  int t   = row % TTGT;
  int tok = target[row];
  int ii  = d >> 1;
  float den = __powf(10000.f, (2.f * (float)ii) / (float)DM);
  float ang = (float)t / den;
  float pe  = (d & 1) ? __cosf(ang) : __sinf(ang);
  float v   = emb[(size_t)tok * DM + d] + pe;
  xf[i]  = v;
  xbf[i] = f2bf(v);
}

// Masked softmax over rows of att[Z*Q, Kd]; one wave32 per row (Kd==512).
__global__ __launch_bounds__(256)
void softmax_mask_kernel(float* __restrict__ att, u16* __restrict__ attbf,
                         const float* __restrict__ qpad, const float* __restrict__ kpad,
                         int Q, int Kd, int causal) {
  const int wave = threadIdx.x >> 5, lane = threadIdx.x & 31;
  const int row  = blockIdx.x * 8 + wave;
  const int q    = row % Q;
  const int zz   = row / Q;
  const int b    = zz / NH;
  float* arow = att + (size_t)row * Kd;
  const float qp = qpad[b * Q + q];
  float vals[16];
  float mx = -3.0e38f;
#pragma unroll
  for (int j = 0; j < 16; ++j) {
    const int c = lane + 32 * j;
    float v = arow[c];
    bool masked = (qp == 0.f) || (kpad[b * Kd + c] == 0.f) || (causal && c > q);
    v = masked ? -1e32f : v;
    vals[j] = v;
    mx = fmaxf(mx, v);
  }
#pragma unroll
  for (int o = 16; o >= 1; o >>= 1) mx = fmaxf(mx, __shfl_xor(mx, o, 32));
  float sum = 0.f;
#pragma unroll
  for (int j = 0; j < 16; ++j) { vals[j] = __expf(vals[j] - mx); sum += vals[j]; }
#pragma unroll
  for (int o = 16; o >= 1; o >>= 1) sum += __shfl_xor(sum, o, 32);
  const float inv = 1.f / sum;
#pragma unroll
  for (int j = 0; j < 16; ++j) {
    const int c = lane + 32 * j;
    const float v = vals[j] * inv;
    arow[c] = v;
    attbf[(size_t)row * Kd + c] = f2bf(v);
  }
}

// x_out = LayerNorm(x + a) * g + b ; one wave32 per row of DM=512
__global__ __launch_bounds__(256)
void add_ln_kernel(const float* __restrict__ x, const float* __restrict__ a,
                   const float* __restrict__ g, const float* __restrict__ be,
                   float* __restrict__ xo, u16* __restrict__ xob) {
  const int wave = threadIdx.x >> 5, lane = threadIdx.x & 31;
  const int row  = blockIdx.x * 8 + wave;
  const float* xr = x + (size_t)row * DM;
  const float* ar = a + (size_t)row * DM;
  float v[16];
  float s = 0.f;
#pragma unroll
  for (int j = 0; j < 16; ++j) {
    const int c = lane + 32 * j;
    v[j] = xr[c] + ar[c];
    s += v[j];
  }
#pragma unroll
  for (int o = 16; o >= 1; o >>= 1) s += __shfl_xor(s, o, 32);
  const float mean = s / (float)DM;
  float qv = 0.f;
#pragma unroll
  for (int j = 0; j < 16; ++j) { const float d = v[j] - mean; qv += d * d; }
#pragma unroll
  for (int o = 16; o >= 1; o >>= 1) qv += __shfl_xor(qv, o, 32);
  const float rstd = rsqrtf(qv / (float)DM + 1e-5f);
#pragma unroll
  for (int j = 0; j < 16; ++j) {
    const int c = lane + 32 * j;
    const float o = (v[j] - mean) * rstd * g[c] + be[c];
    xo[(size_t)row * DM + c]  = o;
    xob[(size_t)row * DM + c] = f2bf(o);
  }
}

// ---------------------------------------------------------------------------
// Host orchestration
// ---------------------------------------------------------------------------
static inline void launch_gemm(hipStream_t st,
    const u16* A, long long lda, long long sAb, long long sAh,
    const u16* Bm, long long ldb, long long sBb, long long sBh, int transB,
    float* C, long long ldc, long long sCb, long long sCh,
    int M, int N, int K, int Z, int batchH,
    const float* bias, float alpha, int relu)
{
  dim3 grid((N + BNT - 1) / BNT, (M + BMT - 1) / BMT, Z);
  gemm_bf16_wmma<<<grid, 256, 0, st>>>(A, lda, sAb, sAh, Bm, ldb, sBb, sBh, transB,
                                       C, ldc, sCb, sCh, M, N, K, batchH,
                                       bias, alpha, relu);
}
static inline void launch_conv(hipStream_t st, const float* s, u16* d, long long n) {
  f32_to_bf16_kernel<<<(unsigned)((n + 255) / 256), 256, 0, st>>>(s, d, n);
}

extern "C" void kernel_launch(void* const* d_in, const int* in_sizes, int n_in,
                              void* d_out, int out_size, void* d_ws, size_t ws_size,
                              hipStream_t stream) {
  (void)in_sizes; (void)n_in; (void)out_size; (void)ws_size;
  const float* encoded_source = (const float*)d_in[0];
  const int*   source_padding = (const int*)d_in[1];
  const int*   target         = (const int*)d_in[2];
  const float* emb            = (const float*)d_in[3];
  const float* Wq1 = (const float*)d_in[4];  const float* bq1 = (const float*)d_in[5];
  const float* Wk1 = (const float*)d_in[6];  const float* bk1 = (const float*)d_in[7];
  const float* Wv1 = (const float*)d_in[8];  const float* bv1 = (const float*)d_in[9];
  const float* Wo1 = (const float*)d_in[10]; const float* bo1 = (const float*)d_in[11];
  const float* Wq2 = (const float*)d_in[12]; const float* bq2 = (const float*)d_in[13];
  const float* Wk2 = (const float*)d_in[14]; const float* bk2 = (const float*)d_in[15];
  const float* Wv2 = (const float*)d_in[16]; const float* bv2 = (const float*)d_in[17];
  const float* Wo2 = (const float*)d_in[18]; const float* bo2 = (const float*)d_in[19];
  const float* ln1_g = (const float*)d_in[20];
  const float* ln2_g = (const float*)d_in[21];
  const float* ln3_g = (const float*)d_in[22];
  const float* ln1_b = (const float*)d_in[23];
  const float* ln2_b = (const float*)d_in[24];
  const float* ln3_b = (const float*)d_in[25];
  const float* W_ff1 = (const float*)d_in[26]; const float* b_ff1 = (const float*)d_in[27];
  const float* W_ff2 = (const float*)d_in[28]; const float* b_ff2 = (const float*)d_in[29];
  const float* lm_W  = (const float*)d_in[30]; const float* lm_b  = (const float*)d_in[31];

  // ---- workspace layout (256B aligned slices)
  char* p = (char*)d_ws;
  auto take = [&](size_t bytes) -> char* {
    char* r = p;
    p += (bytes + 255) & ~(size_t)255;
    return r;
  };
  float* xf    = (float*)take((size_t)BT * DM * 4);
  u16*   xbf   = (u16*)  take((size_t)BT * DM * 2);
  u16*   esbf  = (u16*)  take((size_t)BS * DM * 2);
  u16*   wbf   = (u16*)  take((size_t)DM * NV * 2);   // largest weight (lm_W)
  float* projf = (float*)take((size_t)BT * HD * 4);
  u16*   qbf   = (u16*)  take((size_t)BT * HD * 2);
  u16*   kbf   = (u16*)  take((size_t)BS * HD * 2);
  u16*   vbf   = (u16*)  take((size_t)BS * HD * 2);
  float* attf  = (float*)take((size_t)NB * NH * TTGT * TSRC * 4);
  u16*   attbf = (u16*)  take((size_t)NB * NH * TTGT * TSRC * 2);
  u16*   aobf  = (u16*)  take((size_t)BT * HD * 2);
  float* af    = (float*)take((size_t)BT * DM * 4);
  float* hf    = (float*)take((size_t)BT * FF * 4);
  u16*   hbf   = (u16*)  take((size_t)BT * FF * 2);
  float* tpad  = (float*)take((size_t)BT * 4);
  float* spad  = (float*)take((size_t)BS * 4);

  const float inv_sqrt_d = 1.0f / sqrtf((float)DM);

  // ---- padding masks + embedding + source conversion
  pad_from_tokens_kernel<<<(BT + 255) / 256, 256, 0, stream>>>(target, tpad, BT, VOCAB);
  pad_from_int_kernel<<<(BS + 255) / 256, 256, 0, stream>>>(source_padding, spad, BS);
  embed_kernel<<<(unsigned)(((long long)BT * DM + 255) / 256), 256, 0, stream>>>(
      target, emb, xf, xbf);
  launch_conv(stream, encoded_source, esbf, (long long)BS * DM);

  for (int l = 0; l < NL; ++l) {
    const size_t wOff = (size_t)l * NH * DM * DM;
    const size_t bOff = (size_t)l * NH * DM;

    // ===================== self-attention =====================
    struct { const float *W, *b; u16* dst; } projs1[3] = {
      {Wq1 + wOff, bq1 + bOff, qbf}, {Wk1 + wOff, bk1 + bOff, kbf},
      {Wv1 + wOff, bv1 + bOff, vbf}};
    for (int pidx = 0; pidx < 3; ++pidx) {
      permute_qkv_w_kernel<<<(unsigned)(((long long)DM * HD + 255) / 256), 256, 0, stream>>>(
          projs1[pidx].W, wbf);
      launch_gemm(stream, xbf, DM, 0, 0, wbf, HD, 0, 0, /*NT=*/0,
                  projf, HD, 0, 0, BT, HD, DM, 1, 1, projs1[pidx].b, 1.f, 0);
      launch_conv(stream, projf, projs1[pidx].dst, (long long)BT * HD);
    }
    // scores = Q x K^T / sqrt(D)   (batched over b,h)
    launch_gemm(stream,
                qbf, HD, (long long)TTGT * HD, DM,
                kbf, HD, (long long)TTGT * HD, DM, /*NT=*/1,
                attf, TTGT, (long long)NH * TTGT * TTGT, (long long)TTGT * TTGT,
                TTGT, TTGT, DM, NB * NH, NH, nullptr, inv_sqrt_d, 0);
    softmax_mask_kernel<<<(NB * NH * TTGT) / 8, 256, 0, stream>>>(
        attf, attbf, tpad, tpad, TTGT, TTGT, /*causal=*/1);
    // out = att x V  -> concat heads layout [b,q, h*DM+e]
    launch_gemm(stream,
                attbf, TTGT, (long long)NH * TTGT * TTGT, (long long)TTGT * TTGT,
                vbf, HD, (long long)TTGT * HD, DM, /*NT=*/0,
                projf, HD, (long long)TTGT * HD, DM,
                TTGT, DM, TTGT, NB * NH, NH, nullptr, 1.f, 0);
    launch_conv(stream, projf, aobf, (long long)BT * HD);
    launch_conv(stream, Wo1 + (size_t)l * HD * DM, wbf, (long long)HD * DM);
    launch_gemm(stream, aobf, HD, 0, 0, wbf, DM, 0, 0, 0,
                af, DM, 0, 0, BT, DM, HD, 1, 1, bo1 + (size_t)l * DM, 1.f, 0);
    add_ln_kernel<<<BT / 8, 256, 0, stream>>>(xf, af, ln1_g + (size_t)l * DM,
                                              ln1_b + (size_t)l * DM, xf, xbf);

    // ===================== cross-attention =====================
    // Q from x, K/V from encoded source
    permute_qkv_w_kernel<<<(unsigned)(((long long)DM * HD + 255) / 256), 256, 0, stream>>>(
        Wq2 + wOff, wbf);
    launch_gemm(stream, xbf, DM, 0, 0, wbf, HD, 0, 0, 0,
                projf, HD, 0, 0, BT, HD, DM, 1, 1, bq2 + bOff, 1.f, 0);
    launch_conv(stream, projf, qbf, (long long)BT * HD);
    struct { const float *W, *b; u16* dst; } projs2[2] = {
      {Wk2 + wOff, bk2 + bOff, kbf}, {Wv2 + wOff, bv2 + bOff, vbf}};
    for (int pidx = 0; pidx < 2; ++pidx) {
      permute_qkv_w_kernel<<<(unsigned)(((long long)DM * HD + 255) / 256), 256, 0, stream>>>(
          projs2[pidx].W, wbf);
      launch_gemm(stream, esbf, DM, 0, 0, wbf, HD, 0, 0, 0,
                  projf, HD, 0, 0, BS, HD, DM, 1, 1, projs2[pidx].b, 1.f, 0);
      launch_conv(stream, projf, projs2[pidx].dst, (long long)BS * HD);
    }
    launch_gemm(stream,
                qbf, HD, (long long)TTGT * HD, DM,
                kbf, HD, (long long)TSRC * HD, DM, /*NT=*/1,
                attf, TSRC, (long long)NH * TTGT * TSRC, (long long)TTGT * TSRC,
                TTGT, TSRC, DM, NB * NH, NH, nullptr, inv_sqrt_d, 0);
    softmax_mask_kernel<<<(NB * NH * TTGT) / 8, 256, 0, stream>>>(
        attf, attbf, tpad, spad, TTGT, TSRC, /*causal=*/0);
    if (l == 0) {
      float* att0_dst = (float*)d_out + (size_t)BT * NV;
      (void)hipMemcpyAsync(att0_dst, attf,
                           (size_t)NB * NH * TTGT * TSRC * sizeof(float),
                           hipMemcpyDeviceToDevice, stream);
    }
    launch_gemm(stream,
                attbf, TSRC, (long long)NH * TTGT * TSRC, (long long)TTGT * TSRC,
                vbf, HD, (long long)TSRC * HD, DM, /*NT=*/0,
                projf, HD, (long long)TTGT * HD, DM,
                TTGT, DM, TSRC, NB * NH, NH, nullptr, 1.f, 0);
    launch_conv(stream, projf, aobf, (long long)BT * HD);
    launch_conv(stream, Wo2 + (size_t)l * HD * DM, wbf, (long long)HD * DM);
    launch_gemm(stream, aobf, HD, 0, 0, wbf, DM, 0, 0, 0,
                af, DM, 0, 0, BT, DM, HD, 1, 1, bo2 + (size_t)l * DM, 1.f, 0);
    add_ln_kernel<<<BT / 8, 256, 0, stream>>>(xf, af, ln2_g + (size_t)l * DM,
                                              ln2_b + (size_t)l * DM, xf, xbf);

    // ===================== feed-forward =====================
    launch_conv(stream, W_ff1 + (size_t)l * DM * FF, wbf, (long long)DM * FF);
    launch_gemm(stream, xbf, DM, 0, 0, wbf, FF, 0, 0, 0,
                hf, FF, 0, 0, BT, FF, DM, 1, 1, b_ff1 + (size_t)l * FF, 1.f, /*relu=*/1);
    launch_conv(stream, hf, hbf, (long long)BT * FF);
    launch_conv(stream, W_ff2 + (size_t)l * FF * DM, wbf, (long long)FF * DM);
    launch_gemm(stream, hbf, FF, 0, 0, wbf, DM, 0, 0, 0,
                af, DM, 0, 0, BT, DM, FF, 1, 1, b_ff2 + (size_t)l * DM, 1.f, 0);
    add_ln_kernel<<<BT / 8, 256, 0, stream>>>(xf, af, ln3_g + (size_t)l * DM,
                                              ln3_b + (size_t)l * DM, xf, xbf);
  }

  // ===================== LM head: logits -> d_out =====================
  launch_conv(stream, lm_W, wbf, (long long)DM * NV);
  launch_gemm(stream, xbf, DM, 0, 0, wbf, NV, 0, 0, 0,
              (float*)d_out, NV, 0, 0, BT, NV, DM, 1, 1, lm_b, 1.f, 0);
}
